// _ScaledDotProductAttention_12171937316878
// MI455X (gfx1250) — matrix operand order repaired and checked by reading, not compiled, not run on hardware
//
#include <hip/hip_runtime.h>

// ---------------------------------------------------------------------------
// Scaled-dot-product attention with additive mask + residual bias, returning
// (context, attn, scores).  B=2,H=16,S=2048,Dk=64, fp32 I/O.
// GEMMs run on v_wmma_f32_16x16x32_bf16 (wave32, CDNA5).
// ---------------------------------------------------------------------------

typedef __attribute__((ext_vector_type(16))) __bf16 v16bf;
typedef __attribute__((ext_vector_type(8)))  float  v8f;

constexpr int B_  = 2;
constexpr int H_  = 16;
constexpr int S_  = 2048;
constexpr int DK_ = 64;
constexpr int BH  = B_ * H_;     // 32
constexpr int NT  = S_ / 16;     // 128 column tiles / row tiles

// ---------------------------------------------------------------------------
// Pack kernels: fp32 -> bf16, laid out per WMMA lane in this file's contiguous-K order (same for A and B) so
// the hot kernel issues contiguous 32B-per-lane loads (2 x global_load_b128).
//
// A-layout (16-bit A 16x32): lane l holds row M=l%16; vector elem j maps
//   K = 16*(l/16) + j
// B-layout (16-bit B 32x16): lane l holds col N=l%16; vector elem j maps
//   K = 16*(l/16) + j
// ---------------------------------------------------------------------------

__global__ void pack_q(const float* __restrict__ q, __bf16* __restrict__ qpk) {
  const int gid = blockIdx.x * blockDim.x + threadIdx.x;   // BH*NT*2*32*16
  const int j     =  gid        & 15;
  const int lane  = (gid >> 4)  & 31;
  const int chunk = (gid >> 9)  & 1;
  const int stile = (gid >> 10) & 127;
  const int bh    =  gid >> 17;
  const int g = lane >> 4, m = lane & 15;
  const int v = j >> 1,   h = j & 1;
  const int K = 16 * g + 2 * v + h;
  const int s = stile * 16 + m;
  const int d = chunk * 32 + K;
  qpk[gid] = (__bf16)q[((size_t)bh * S_ + s) * DK_ + d];
}

__global__ void pack_k(const float* __restrict__ k, __bf16* __restrict__ kpk) {
  const int gid = blockIdx.x * blockDim.x + threadIdx.x;   // BH*NT*2*32*16
  const int j     =  gid        & 15;
  const int lane  = (gid >> 4)  & 31;
  const int chunk = (gid >> 9)  & 1;
  const int tcol  = (gid >> 10) & 127;
  const int bh    =  gid >> 17;
  const int d = chunk * 32 + 16 * (lane >> 4) + j;
  const int t = tcol * 16 + (lane & 15);
  kpk[gid] = (__bf16)k[((size_t)bh * DK_ + d) * S_ + t];
}

__global__ void pack_v(const float* __restrict__ v, __bf16* __restrict__ vpk) {
  const int gid = blockIdx.x * blockDim.x + threadIdx.x;   // BH*64*4*32*16
  const int j     =  gid        & 15;
  const int lane  = (gid >> 4)  & 31;
  const int ntile = (gid >> 9)  & 3;
  const int tpair = (gid >> 11) & 63;
  const int bh    =  gid >> 17;
  const int t = tpair * 32 + 16 * (lane >> 4) + j;
  const int d = ntile * 16 + (lane & 15);
  vpk[gid] = (__bf16)v[((size_t)bh * S_ + t) * DK_ + d];
}

// ---------------------------------------------------------------------------
// Main fused kernel: one wave32 per 16-row tile of one (b,h) head.
// ---------------------------------------------------------------------------
__global__ __launch_bounds__(256, 1)
void attn_fused(const float* __restrict__ prev, const float* __restrict__ mask,
                const __bf16* __restrict__ qpk, const __bf16* __restrict__ kpk,
                const __bf16* __restrict__ vpk,
                float* __restrict__ ctx, float* __restrict__ attn,
                float* __restrict__ scores)
{
  __shared__ float lds[8][16 * 34];          // stride 34: conflict-free, 8B aligned pairs

  const int lane  = threadIdx.x & 31;
  const int wv    = threadIdx.x >> 5;
  const int tile  = blockIdx.x * 8 + wv;     // 0..4095
  const int bh    = tile >> 7;
  const int stile = tile & 127;
  const int g     = lane >> 4;               // 16-lane half (C-layout M split)
  const int n     = lane & 15;               // C/B-layout column index

  // Q tile, A-layout bf16, K = 0..63 in two chunks of 32
  const v16bf qa0 = *(const v16bf*)(qpk + (((size_t)tile * 2 + 0) * 32 + lane) * 16);
  const v16bf qa1 = *(const v16bf*)(qpk + (((size_t)tile * 2 + 1) * 32 + lane) * 16);

  const size_t bhSS = (size_t)bh * S_ * S_;
  const float  scale = 0.125f;               // 1/sqrt(64)

  float smax[8];
#pragma unroll
  for (int r = 0; r < 8; ++r) smax[r] = -3.0e38f;

  // -------- Pass 1: scores = QK^T/8 + mask + prev, track row max -----------
  for (int tc = 0; tc < NT; ++tc) {
    const v16bf kb0 = *(const v16bf*)(kpk + ((((size_t)bh * NT + tc) * 2 + 0) * 32 + lane) * 16);
    const v16bf kb1 = *(const v16bf*)(kpk + ((((size_t)bh * NT + tc) * 2 + 1) * 32 + lane) * 16);
    v8f acc = {};
    acc = __builtin_amdgcn_wmma_f32_16x16x32_bf16(false, qa0, false, kb0, (short)0, acc, false, false);
    acc = __builtin_amdgcn_wmma_f32_16x16x32_bf16(false, qa1, false, kb1, (short)0, acc, false, false);
    const int t = tc * 16 + n;
#pragma unroll
    for (int r = 0; r < 8; ++r) {
      const int    s   = stile * 16 + r + 8 * g;
      const size_t idx = bhSS + (size_t)s * S_ + t;
      const float  sc  = acc[r] * scale + mask[(size_t)s * S_ + t] + prev[idx];
      scores[idx] = sc;
      smax[r] = fmaxf(smax[r], sc);
    }
  }
#pragma unroll
  for (int r = 0; r < 8; ++r) {              // reduce across each 16-lane half
    float m = smax[r];
    m = fmaxf(m, __shfl_xor(m, 1, 32));
    m = fmaxf(m, __shfl_xor(m, 2, 32));
    m = fmaxf(m, __shfl_xor(m, 4, 32));
    m = fmaxf(m, __shfl_xor(m, 8, 32));
    smax[r] = m;
  }

  // -------- Pass 2: exp + row sums (scores re-read hits L2) ----------------
  float rinv[8];
#pragma unroll
  for (int r = 0; r < 8; ++r) rinv[r] = 0.0f;
  for (int tc = 0; tc < NT; ++tc) {
    const int t = tc * 16 + n;
#pragma unroll
    for (int r = 0; r < 8; ++r) {
      const int    s   = stile * 16 + r + 8 * g;
      const size_t idx = bhSS + (size_t)s * S_ + t;
      const float  e   = __expf(scores[idx] - smax[r]);
      rinv[r] += e;
      attn[idx] = e;                         // stash unnormalized exp
    }
  }
#pragma unroll
  for (int r = 0; r < 8; ++r) {
    float sb = rinv[r];
    sb += __shfl_xor(sb, 1, 32);
    sb += __shfl_xor(sb, 2, 32);
    sb += __shfl_xor(sb, 4, 32);
    sb += __shfl_xor(sb, 8, 32);
    rinv[r] = __builtin_amdgcn_rcpf(sb);
  }

  // -------- Pass 3: normalize attn, LDS C->A transpose, context GEMM -------
  v8f c0 = {}, c1 = {}, c2 = {}, c3 = {};
  float* myLds = &lds[wv][0];
  for (int tp = 0; tp < NT / 2; ++tp) {      // K=32 chunks of the P matrix
#pragma unroll
    for (int sub = 0; sub < 2; ++sub) {
      const int tc = tp * 2 + sub;
      const int t  = tc * 16 + n;
#pragma unroll
      for (int r = 0; r < 8; ++r) {
        const int    s   = stile * 16 + r + 8 * g;
        const size_t idx = bhSS + (size_t)s * S_ + t;
        const float  p   = attn[idx] * rinv[r];
        attn[idx] = p;                                   // final attn output
        myLds[(r + 8 * g) * 34 + sub * 16 + n] = p;      // C-layout -> LDS
      }
    }
    asm volatile("s_wait_dscnt 0" ::: "memory");         // cross-lane DS hazard

    v16bf pa{};                                          // A-layout gather
#pragma unroll
    for (int vv = 0; vv < 8; ++vv) {
      const int   K0 = 16 * g + 2 * vv;
      const float2 f2 = *(const float2*)(myLds + n * 34 + K0);
      pa[2 * vv]     = (__bf16)f2.x;
      pa[2 * vv + 1] = (__bf16)f2.y;
    }

    const __bf16* vb = vpk + (((size_t)bh * 64 + tp) * 4) * 32 * 16;
    const v16bf vb0 = *(const v16bf*)(vb + (0 * 32 + lane) * 16);
    const v16bf vb1 = *(const v16bf*)(vb + (1 * 32 + lane) * 16);
    const v16bf vb2 = *(const v16bf*)(vb + (2 * 32 + lane) * 16);
    const v16bf vb3 = *(const v16bf*)(vb + (3 * 32 + lane) * 16);
    c0 = __builtin_amdgcn_wmma_f32_16x16x32_bf16(false, pa, false, vb0, (short)0, c0, false, false);
    c1 = __builtin_amdgcn_wmma_f32_16x16x32_bf16(false, pa, false, vb1, (short)0, c1, false, false);
    c2 = __builtin_amdgcn_wmma_f32_16x16x32_bf16(false, pa, false, vb2, (short)0, c2, false, false);
    c3 = __builtin_amdgcn_wmma_f32_16x16x32_bf16(false, pa, false, vb3, (short)0, c3, false, false);
  }

  // context out: 16 rows x 64 cols, C-layout
#pragma unroll
  for (int r = 0; r < 8; ++r) {
    const int s = stile * 16 + r + 8 * g;
    float* cp = ctx + ((size_t)bh * S_ + s) * DK_;
    cp[0 * 16 + n] = c0[r];
    cp[1 * 16 + n] = c1[r];
    cp[2 * 16 + n] = c2[r];
    cp[3 * 16 + n] = c3[r];
  }
}

// ---------------------------------------------------------------------------
extern "C" void kernel_launch(void* const* d_in, const int* in_sizes, int n_in,
                              void* d_out, int out_size, void* d_ws, size_t ws_size,
                              hipStream_t stream) {
  const float* q    = (const float*)d_in[0];
  const float* k    = (const float*)d_in[1];
  const float* v    = (const float*)d_in[2];
  const float* prev = (const float*)d_in[3];
  const float* mask = (const float*)d_in[4];

  float* out    = (float*)d_out;
  float* ctx    = out;                                  // B*H*S*DK
  float* attn   = out + (size_t)BH * S_ * DK_;          // B*H*S*S
  float* scores = attn + (size_t)BH * S_ * S_;          // B*H*S*S

  __bf16* qpk = (__bf16*)d_ws;                          // 8 MB each
  __bf16* kpk = qpk + (size_t)BH * S_ * DK_;
  __bf16* vpk = kpk + (size_t)BH * S_ * DK_;

  const int nelem = BH * S_ * DK_;                      // 4,194,304
  pack_q<<<nelem / 256, 256, 0, stream>>>(q, qpk);
  pack_k<<<nelem / 256, 256, 0, stream>>>(k, kpk);
  pack_v<<<nelem / 256, 256, 0, stream>>>(v, vpk);

  attn_fused<<<(BH * NT) / 8, 256, 0, stream>>>(prev, mask, qpk, kpk, vpk,
                                                ctx, attn, scores);
}
